// MultiHashEmbedding_26293789786878
// MI455X (gfx1250) — matrix operand order repaired and checked by reading, not compiled, not run on hardware
//
#include <hip/hip_runtime.h>

typedef __attribute__((ext_vector_type(16))) __bf16 v16bf;
typedef __attribute__((ext_vector_type(8)))  float  v8f;
typedef int v4i_ __attribute__((vector_size(16)));

#define D_MODEL 1024
#define FAN_IN  1024

// d_ws layout (bytes): [0,2M) packed W bf16 | [2M, 2M+4.9M) bf16 tables | [7M, +64K) sums
#define WS_WPACK_OFF  0
#define WS_TAB_OFF    (2u * 1024u * 1024u)
#define WS_SUMS_OFF   (7u * 1024u * 1024u)

#define SCHED_FENCE() __builtin_amdgcn_sched_barrier(0)

#if __has_builtin(__builtin_amdgcn_global_load_async_to_lds_b128)
#define HAVE_ASYNC_LDS 1
// per-lane 16B global -> LDS copy, tracked by ASYNCcnt (no VGPR round-trip)
static __device__ __forceinline__ void async_cp16(const void* g, void* l) {
    __builtin_amdgcn_global_load_async_to_lds_b128((v4i_*)g, (v4i_*)l, 0, 0);
}
#endif

// fp32 -> bf16 round-to-nearest-even
static __device__ __forceinline__ unsigned int f2bf(float f) {
    unsigned int u = __float_as_uint(f);
    u += 0x7FFFu + ((u >> 16) & 1u);
    return u >> 16;
}
static __device__ __forceinline__ unsigned int pk2(float lo, float hi) {
    return f2bf(lo) | (f2bf(hi) << 16);
}

// ---------------------------------------------------------------------------
// Pre-pass 1: fusion_w [1024 d][1024 f] fp32 -> bf16 in WMMA B-fragment lane
// layout.  Block blk = ntg*32 + kt; lane L holds col d = ntg*16 + (L&15),
// sixteen consecutive k values f = kt*32 + (L>>4)*16.
// ---------------------------------------------------------------------------
__global__ __launch_bounds__(256) void pack_w_bf16(const float* __restrict__ W,
                                                   unsigned int* __restrict__ wp) {
    int g    = blockIdx.x * blockDim.x + threadIdx.x;   // 65536 threads
    int blk  = g >> 5;
    int lane = g & 31;
    int ntg  = blk >> 5;
    int kt   = blk & 31;
    int d    = ntg * 16 + (lane & 15);
    int f    = kt * 32 + (lane >> 4) * 16;
    const float* src = W + (size_t)d * FAN_IN + f;
    unsigned int* dst = wp + (size_t)g * 8;
#pragma unroll
    for (int i = 0; i < 8; ++i)
        dst[i] = pk2(src[2 * i], src[2 * i + 1]);
}

// ---------------------------------------------------------------------------
// Pre-pass 2: hash tables fp32 -> bf16 (same row-major layout).
// ---------------------------------------------------------------------------
__global__ __launch_bounds__(256) void pack_tables(
    const float* t0, const float* t1, const float* t2, const float* t3,
    const float* t4, const float* t5, const float* t6, const float* t7,
    unsigned short* __restrict__ dst) {
    int j = blockIdx.y;
    const float* s;
    int n, off;
    switch (j) {
    case 0:  s = t0; n = 251  * 128; off = 0;       break;
    case 1:  s = t1; n = 509  * 128; off = 32128;   break;
    case 2:  s = t2; n = 1021 * 128; off = 97280;   break;
    case 3:  s = t3; n = 2039 * 128; off = 227968;  break;
    case 4:  s = t4; n = 4093 * 128; off = 488960;  break;
    case 5:  s = t5; n = 8191 * 128; off = 1012864; break;
    case 6:  s = t6; n = 997  * 128; off = 2061312; break;
    default: s = t7; n = 1999 * 128; off = 2188928; break;
    }
    int i4 = (blockIdx.x * 256 + threadIdx.x) * 4;
    if (i4 < n) {
        float4 v = *(const float4*)(s + i4);
        unsigned int* d = (unsigned int*)(dst + off + i4);
        d[0] = pk2(v.x, v.y);
        d[1] = pk2(v.z, v.w);
    }
}

__global__ __launch_bounds__(256) void zero_f32(float* p, int n) {
    int i = blockIdx.x * 256 + threadIdx.x;
    if (i < n) p[i] = 0.0f;
}

// ---------------------------------------------------------------------------
// K1: gather + bf16 WMMA GEMM + bias; writes un-normalized y and atomically
// accumulates per-token sum(y^2).
// Grid (128 M-blocks, 8 N-slices) x 256 threads (8 waves).
// WG = 128 tokens x 128 cols; per-kt A/B slices staged to LDS, double
// buffered via GLOBAL_LOAD_ASYNC_TO_LDS_B128 (ASYNCcnt).  kt loop pinned at
// unroll 1: one iteration's fragments live -> no accumulator copies.
// ---------------------------------------------------------------------------
__global__ __launch_bounds__(256) void gemm_partial(
    const int* __restrict__ tok,
    const unsigned short* __restrict__ tabbf,
    const unsigned int* __restrict__ wp,
    const float* __restrict__ bias,
    float* __restrict__ sums,
    float* __restrict__ out) {

    __shared__ unsigned short Ab[2][4096];   // 128 tok x 32 k, fragment layout
    __shared__ unsigned short Bb[2][4096];   // 8 tiles x 32 lanes x 16 bf16
    __shared__ int bk[8 * 128];              // bucket per (table, token)
    __shared__ int toff[8];                  // bf16 table base offsets

    const int tid     = threadIdx.x;
    const int Mbase   = blockIdx.x * 128;
    const int nsl     = blockIdx.y;          // cols [nsl*128, +128)
    const int ntgBase = nsl * 8;

    if (tid < 8) {
        const int TOFF[8] = {0, 32128, 97280, 227968, 488960, 1012864, 2061312, 2188928};
        toff[tid] = TOFF[tid];
    }
    if (tid < 128) {
        int id = tok[Mbase + tid];
        const int P[8] = {251, 509, 1021, 2039, 4093, 8191, 997, 1999};
#pragma unroll
        for (int j = 0; j < 8; ++j) bk[j * 128 + tid] = id % P[j];
    }
    __syncthreads();

    // per-thread staging roles (constant)
    const int sm  = tid >> 1, sp = tid & 1;   // A: token, 16-k half-chunk
    const int sw  = sm >> 4, slm = sm & 15;
    const int bnt = tid >> 5, bln = tid & 31; // B: n-tile, lane

#if HAVE_ASYNC_LDS
    auto stage = [&](int kt, int buf) {
        int j = kt >> 2;
        int bucket = bk[j * 128 + sm];
        const uint4* asrc = (const uint4*)(tabbf + toff[j] +
                                           (size_t)bucket * 128 + (kt & 3) * 32 + sp * 16);
        const uint4* bsrc = (const uint4*)wp +
                            ((size_t)(((ntgBase + bnt) * 32 + kt) * 32 + bln)) * 2;
        uint4* Ad = (uint4*)Ab[buf];
        uint4* Bd = (uint4*)Bb[buf];
        async_cp16(asrc,     &Ad[(sw * 32 + slm + 16 * sp) * 2]);
        async_cp16(asrc + 1, &Ad[(sw * 32 + slm + 16 * sp) * 2 + 1]);
        async_cp16(bsrc,     &Bd[tid * 2]);
        async_cp16(bsrc + 1, &Bd[tid * 2 + 1]);
    };
    auto stage_commit = [&]() {
        asm volatile("s_wait_asynccnt 0x0" ::: "memory");
    };
#else
    uint4 a0, a1, b0, b1;
    int pend_buf = 0;
    auto stage = [&](int kt, int buf) {
        pend_buf = buf;
        int j = kt >> 2;
        int bucket = bk[j * 128 + sm];
        const uint4* asrc = (const uint4*)(tabbf + toff[j] +
                                           (size_t)bucket * 128 + (kt & 3) * 32 + sp * 16);
        a0 = asrc[0];
        a1 = asrc[1];
        const uint4* bsrc = (const uint4*)wp +
                            ((size_t)(((ntgBase + bnt) * 32 + kt) * 32 + bln)) * 2;
        b0 = bsrc[0];
        b1 = bsrc[1];
    };
    auto stage_commit = [&]() {
        uint4* Ad = (uint4*)Ab[pend_buf];
        Ad[(sw * 32 + slm + 16 * sp) * 2]     = a0;
        Ad[(sw * 32 + slm + 16 * sp) * 2 + 1] = a1;
        uint4* Bd = (uint4*)Bb[pend_buf];
        Bd[tid * 2]     = b0;
        Bd[tid * 2 + 1] = b1;
    };
#endif

    stage(0, 0);
    stage_commit();
    __syncthreads();

    const int lane = tid & 31, wv = tid >> 5;
    const int lm = lane & 15, half = lane >> 4;

    v8f acc[8];
#pragma unroll
    for (int n = 0; n < 8; ++n)
#pragma unroll
        for (int i = 0; i < 8; ++i) acc[n][i] = 0.0f;

#pragma clang loop unroll(disable)
    for (int kt = 0; kt < 32; ++kt) {
        int cur = kt & 1;

        // (1) kick off next-step staging; latency overlaps the body below.
        if (kt < 31) stage(kt + 1, cur ^ 1);
        SCHED_FENCE();

        // (2) one clause of 18 ds_load_b128: A fragment + all 8 B fragments.
        union { uint4 u[2]; v16bf v; } af, bfr[8];
        const uint4* Ap = (const uint4*)Ab[cur] + (wv * 32 + lane) * 2;
        af.u[0] = Ap[0];
        af.u[1] = Ap[1];
        const uint4* Bp = (const uint4*)Bb[cur] + lane * 2;
#pragma unroll
        for (int nt = 0; nt < 8; ++nt) {
            bfr[nt].u[0] = Bp[nt * 64];
            bfr[nt].u[1] = Bp[nt * 64 + 1];
        }
        SCHED_FENCE();

        // (3) 8 back-to-back WMMAs; waits decrement across the load clause.
#pragma unroll
        for (int nt = 0; nt < 8; ++nt)
            acc[nt] = __builtin_amdgcn_wmma_f32_16x16x32_bf16(
                false, af.v, false, bfr[nt].v, (short)0, acc[nt], false, false);

        // (4) wait for staged data for kt+1, then sync the workgroup.
        if (kt < 31) stage_commit();
        __syncthreads();
    }

    // ---- epilogue: +bias, partial sum(y^2) -> global atomics, store y ----
    float bia[8];
#pragma unroll
    for (int nt = 0; nt < 8; ++nt) bia[nt] = bias[nsl * 128 + nt * 16 + lm];

    float part[8];
#pragma unroll
    for (int r = 0; r < 8; ++r) {
        float p = 0.0f;
#pragma unroll
        for (int nt = 0; nt < 8; ++nt) {
            float yv = acc[nt][r] + bia[nt];
            acc[nt][r] = yv;
            p += yv * yv;
        }
        part[r] = p;
    }
#pragma unroll
    for (int r = 0; r < 8; ++r) {
        float p = part[r];
        p += __shfl_xor(p, 1, 16);
        p += __shfl_xor(p, 2, 16);
        p += __shfl_xor(p, 4, 16);
        p += __shfl_xor(p, 8, 16);
        if (lm == 0) atomicAdd(&sums[Mbase + wv * 16 + 8 * half + r], p);
    }

    const int token0 = Mbase + wv * 16 + 8 * half;
    float* obase = out + (size_t)token0 * D_MODEL + nsl * 128 + lm;
#pragma unroll
    for (int r = 0; r < 8; ++r) {
        float* op = obase + (size_t)r * D_MODEL;
#pragma unroll
        for (int nt = 0; nt < 8; ++nt) op[nt * 16] = acc[nt][r];
    }
}

// ---------------------------------------------------------------------------
// K2: y *= rsqrt(mean(y^2)+eps) * gain, in place.  4 floats / thread.
// ---------------------------------------------------------------------------
__global__ __launch_bounds__(256) void rmsnorm_scale(
    float* __restrict__ y, const float* __restrict__ sums,
    const float* __restrict__ gw) {
    int t = blockIdx.x * 256 + threadIdx.x;
    int e = t * 4;
    int row = e >> 10, d = e & 1023;
    float inv = rsqrtf(sums[row] * (1.0f / (float)D_MODEL) + 1e-6f);
    float4 v = *(float4*)(y + e);
    float4 g = *(const float4*)(gw + d);
    v.x *= inv * g.x;
    v.y *= inv * g.y;
    v.z *= inv * g.z;
    v.w *= inv * g.w;
    *(float4*)(y + e) = v;
}

// ---------------------------------------------------------------------------
extern "C" void kernel_launch(void* const* d_in, const int* in_sizes, int n_in,
                              void* d_out, int out_size, void* d_ws, size_t ws_size,
                              hipStream_t stream) {
    const int* tok = (const int*)d_in[0];
    const float* tabs[8];
    const float *W, *bias, *gw;

    if (n_in >= 12) {
        for (int j = 0; j < 8; ++j) tabs[j] = (const float*)d_in[1 + j];
        W    = (const float*)d_in[9];
        bias = (const float*)d_in[10];
        gw   = (const float*)d_in[11];
    } else {
        const int P[8] = {251, 509, 1021, 2039, 4093, 8191, 997, 1999};
        const float* base = (const float*)d_in[1];
        size_t off = 0;
        for (int j = 0; j < 8; ++j) { tabs[j] = base + off; off += (size_t)P[j] * 128; }
        W    = (const float*)d_in[2];
        bias = (const float*)d_in[3];
        gw   = (const float*)d_in[4];
    }

    unsigned char* ws = (unsigned char*)d_ws;
    unsigned int*   wp    = (unsigned int*)(ws + WS_WPACK_OFF);   // 2 MB
    unsigned short* tabbf = (unsigned short*)(ws + WS_TAB_OFF);   // 4.9 MB
    float*          sums  = (float*)(ws + WS_SUMS_OFF);           // 64 KB
    float*          y     = (float*)d_out;

    pack_w_bf16<<<256, 256, 0, stream>>>(W, wp);
    pack_tables<<<dim3(1024, 8), 256, 0, stream>>>(
        tabs[0], tabs[1], tabs[2], tabs[3], tabs[4], tabs[5], tabs[6], tabs[7], tabbf);
    zero_f32<<<64, 256, 0, stream>>>(sums, 16384);

    gemm_partial<<<dim3(128, 8), 256, 0, stream>>>(tok, tabbf, wp, bias, sums, y);

    rmsnorm_scale<<<16384, 256, 0, stream>>>(y, sums, gw);
}
